// CFDGCN_80633716015489
// MI455X (gfx1250) — matrix operand order repaired and checked by reading, not compiled, not run on hardware
//
#include <hip/hip_runtime.h>
#include <hip/hip_bf16.h>
#include <stdint.h>

#define NBATCH 4
#define NFINE  20000
#define NCRS   2000
#define HDIM   512
#define DIN    5
#define OUTC   3
#define NTOT   (NBATCH * NFINE)     // 80000 nodes
#define NEDGE  (NBATCH * 6 * NFINE) // 480000 directed edges (self loops handled separately)
#define K0PAD  32                   // 6  -> 32  (pad K to multiple of 32 for bf16 WMMA)
#define KENDP  544                  // 515 -> 544
#define N2PAD  16                   // 3 -> 16 output cols for last GEMM

typedef __attribute__((ext_vector_type(16))) __bf16 v16bf;
typedef __attribute__((ext_vector_type(8)))  __bf16 v8bf;
typedef __attribute__((ext_vector_type(8)))  float  v8f;

__device__ __forceinline__ __bf16 f2bf(float f) {
    // round-to-nearest-even f32 -> bf16 via integer bits
    uint32_t u = __builtin_bit_cast(uint32_t, f);
    uint32_t r = u + 0x7FFFu + ((u >> 16) & 1u);
    uint16_t s = (uint16_t)(r >> 16);
    return __builtin_bit_cast(__bf16, s);
}

// ---------------- degree / normalization ----------------
__global__ void k_deg_init(float* __restrict__ deg, int n) {
    int i = blockIdx.x * blockDim.x + threadIdx.x;
    if (i < n) deg[i] = 1.0f;  // self loop
}

__global__ void k_deg_edges(const int* __restrict__ col, float* __restrict__ deg, int E) {
    int e = blockIdx.x * blockDim.x + threadIdx.x;
    if (e < E) atomicAdd(&deg[col[e]], 1.0f);
}

__global__ void k_dinv(const float* __restrict__ deg, float* __restrict__ dinv, int n) {
    int i = blockIdx.x * blockDim.x + threadIdx.x;
    if (i < n) dinv[i] = rsqrtf(deg[i]);
}

// ---------------- bf16 input prep ----------------
__global__ void k_build_in0(const float* __restrict__ x, const float* __restrict__ sdf,
                            __bf16* __restrict__ A, int n) {
    int i = blockIdx.x * blockDim.x + threadIdx.x;
    if (i >= n) return;
    __bf16* p = A + (size_t)i * K0PAD;
#pragma unroll
    for (int c = 0; c < DIN; ++c) p[c] = f2bf(x[(size_t)i * DIN + c]);
    p[DIN] = f2bf(sdf[i % NFINE]);
#pragma unroll
    for (int c = DIN + 1; c < K0PAD; ++c) p[c] = f2bf(0.0f);
}

// ---------------- pack W[K,N] (f32, row-major) into WMMA B-fragment layout ----------------
// fragment (nt, kt): 32 lanes x 16 bf16 contiguous:
//   Bp[((nt*Ktiles + kt)*32 + lane)*16 + i] = W[kt*32 + (lane<16?0:16) + i, nt*16 + (lane&15)]
__global__ void k_pack_b(const float* __restrict__ W, __bf16* __restrict__ Bp,
                         int K, int N, int Kpad, int Npad) {
    long long idx = (long long)blockIdx.x * blockDim.x + threadIdx.x;
    if (idx >= (long long)Kpad * Npad) return;
    int i    = (int)(idx & 15);
    int lane = (int)((idx >> 4) & 31);
    long long rest = idx >> 9;
    int Ktiles = Kpad >> 5;
    int kt = (int)(rest % Ktiles);
    int nt = (int)(rest / Ktiles);
    int k  = (kt << 5) + ((lane < 16) ? 0 : 16) + i;
    int nn = (nt << 4) + (lane & 15);
    float v = (k < K && nn < N) ? W[(size_t)k * N + nn] : 0.0f;
    Bp[idx] = f2bf(v);
}

// ---------------- WMMA GEMM ----------------
// C[M,ldc] = A[M,lda](bf16) @ W (packed fragments), f32 accumulate.
// Each wave computes a 16 x (16*NT) strip: A fragment loaded once per K-step,
// reused across NT packed-B fragments (all loads are 16B vector loads).
template <int NT>
__global__ void k_gemm_bf16(const __bf16* __restrict__ A, const __bf16* __restrict__ Bp,
                            float* __restrict__ C, int M, int Ktiles, int N, int lda, int ldc) {
    const int lane = threadIdx.x & 31;
    const int wave = threadIdx.x >> 5;
    const int m0  = blockIdx.x << 4;
    const int nt0 = ((blockIdx.y * (blockDim.x >> 5)) + wave) * NT;
    const int n0  = nt0 << 4;
    if (n0 >= N) return;  // wave-uniform: EXEC stays all-ones for WMMA

    const int l15 = lane & 15;
    const int m   = m0 + l15;
    const int kba = (lane < 16) ? 0 : 16;  // A fragment: lanes 0-15 hold K {0..15}

    const __bf16* ap = A + (size_t)m * lda + kba;
    const __bf16* bp = Bp + (((size_t)nt0 * Ktiles) << 9) + (lane << 4);

    v8f acc[NT] = {};
    for (int kt = 0; kt < Ktiles; ++kt) {
        __builtin_prefetch(ap + 256, 0, 1);  // global_prefetch ahead on A stream
        v8bf alo = *(const v8bf*)(ap);       // 16B aligned
        v8bf ahi = *(const v8bf*)(ap + 8);
        ap += 32;
        v16bf a;
#pragma unroll
        for (int i = 0; i < 8; ++i) { a[i] = alo[i]; a[8 + i] = ahi[i]; }

#pragma unroll
        for (int t = 0; t < NT; ++t) {
            const __bf16* bt = bp + (((size_t)t * Ktiles) << 9);
            v8bf blo = *(const v8bf*)(bt);
            v8bf bhi = *(const v8bf*)(bt + 8);
            v16bf b;
#pragma unroll
            for (int i = 0; i < 8; ++i) { b[i] = blo[i]; b[8 + i] = bhi[i]; }
            acc[t] = __builtin_amdgcn_wmma_f32_16x16x32_bf16(
                false, a, false, b, (short)0, acc[t], false, false);
        }
        bp += 512;  // next K-tile fragment (1 KB stride)
    }

    // C/D layout: lanes 0-15 -> M = m0+r, lanes 16-31 -> M = m0+8+r; N = n0 + (lane&15)
    const int mb = m0 + ((lane < 16) ? 0 : 8);
#pragma unroll
    for (int t = 0; t < NT; ++t) {
        float* crow = C + (size_t)mb * ldc + n0 + (t << 4) + l15;
#pragma unroll
        for (int r = 0; r < 8; ++r) crow[(size_t)r * ldc] = acc[t][r];
    }
}

// ---------------- GCN message passing (f32) ----------------
__global__ void k_self_loop(const float* __restrict__ h, const float* __restrict__ dinv,
                            float* __restrict__ acc, int n, int C4) {
    long long idx = (long long)blockIdx.x * blockDim.x + threadIdx.x;
    if (idx >= (long long)n * C4) return;
    int i = (int)(idx / C4);
    int c = (int)(idx % C4) * 4;
    int C = C4 * 4;
    float d = dinv[i];
    float s = d * d;
    const float4 v = *(const float4*)(h + (size_t)i * C + c);
    float4 o; o.x = s * v.x; o.y = s * v.y; o.z = s * v.z; o.w = s * v.w;
    *(float4*)(acc + (size_t)i * C + c) = o;
}

__global__ void k_scatter_edges(const float* __restrict__ h, const int* __restrict__ row,
                                const int* __restrict__ col, const float* __restrict__ dinv,
                                float* __restrict__ acc, int E, int C4) {
    long long idx = (long long)blockIdx.x * blockDim.x + threadIdx.x;
    if (idx >= (long long)E * C4) return;
    int e = (int)(idx / C4);
    int c = (int)(idx % C4) * 4;
    int C = C4 * 4;
    int r = row[e], cl = col[e];
    float norm = dinv[r] * dinv[cl];
    const float4 v = *(const float4*)(h + (size_t)r * C + c);
    float* dst = acc + (size_t)cl * C + c;
    atomicAdd(dst + 0, norm * v.x);
    atomicAdd(dst + 1, norm * v.y);
    atomicAdd(dst + 2, norm * v.z);
    atomicAdd(dst + 3, norm * v.w);
}

__global__ void k_bias_relu_bf16(const float* __restrict__ acc, const float* __restrict__ bias,
                                 __bf16* __restrict__ out, int n, int C, int ldo, int coff) {
    long long idx = (long long)blockIdx.x * blockDim.x + threadIdx.x;
    if (idx >= (long long)n * C) return;
    int i = (int)(idx / C);
    int c = (int)(idx % C);
    float v = acc[(size_t)i * C + c] + bias[c];
    v = fmaxf(v, 0.0f);
    out[(size_t)i * ldo + coff + c] = f2bf(v);
}

// ---------------- KNN interpolation (k=3) ----------------
__global__ void k_knn(const float* __restrict__ x, const float* __restrict__ cx,
                      const float* __restrict__ cy, float* __restrict__ fy, int n) {
    __shared__ float scx[NCRS * 2];  // 16 KB coarse-point table, LDS-resident
    for (int j = threadIdx.x; j < NCRS * 2; j += blockDim.x) scx[j] = cx[j];
    __syncthreads();
    int i = blockIdx.x * blockDim.x + threadIdx.x;
    if (i >= n) return;
    float px = x[(size_t)i * DIN + 0];
    float py = x[(size_t)i * DIN + 1];
    float d0 = 3.4e38f, d1 = 3.4e38f, d2 = 3.4e38f;
    int j0 = 0, j1 = 0, j2 = 0;
    for (int j = 0; j < NCRS; ++j) {
        float dx = px - scx[2 * j];
        float dy = py - scx[2 * j + 1];
        float d = dx * dx + dy * dy;
        if (d < d0)      { d2 = d1; j2 = j1; d1 = d0; j1 = j0; d0 = d; j0 = j; }
        else if (d < d1) { d2 = d1; j2 = j1; d1 = d;  j1 = j; }
        else if (d < d2) { d2 = d;  j2 = j; }
    }
    float w0 = 1.0f / fmaxf(d0, 1e-16f);
    float w1 = 1.0f / fmaxf(d1, 1e-16f);
    float w2 = 1.0f / fmaxf(d2, 1e-16f);
    float wsum = w0 + w1 + w2;
    const float* cb = cy + (size_t)(i / NFINE) * NCRS * OUTC;
#pragma unroll
    for (int c = 0; c < OUTC; ++c) {
        float v = w0 * cb[(size_t)j0 * OUTC + c] + w1 * cb[(size_t)j1 * OUTC + c] +
                  w2 * cb[(size_t)j2 * OUTC + c];
        fy[(size_t)i * OUTC + c] = v / wsum;
    }
}

// fill concat cols 0..2 with fine_y (bf16) and zero the K padding 515..543
__global__ void k_concat_fy(const float* __restrict__ fy, __bf16* __restrict__ buf, int n) {
    int i = blockIdx.x * blockDim.x + threadIdx.x;
    if (i >= n) return;
    __bf16* p = buf + (size_t)i * KENDP;
    p[0] = f2bf(fy[(size_t)i * OUTC + 0]);
    p[1] = f2bf(fy[(size_t)i * OUTC + 1]);
    p[2] = f2bf(fy[(size_t)i * OUTC + 2]);
#pragma unroll
    for (int c = OUTC + HDIM; c < KENDP; ++c) p[c] = f2bf(0.0f);
}

__global__ void k_final(const float* __restrict__ acc16, const float* __restrict__ b2,
                        float* __restrict__ out, int n) {
    long long idx = (long long)blockIdx.x * blockDim.x + threadIdx.x;
    if (idx >= (long long)n * OUTC) return;
    int i = (int)(idx / OUTC);
    int c = (int)(idx % OUTC);
    out[idx] = acc16[(size_t)i * N2PAD + c] + b2[c];
}

// ---------------- orchestration ----------------
extern "C" void kernel_launch(void* const* d_in, const int* in_sizes, int n_in,
                              void* d_out, int out_size, void* d_ws, size_t ws_size,
                              hipStream_t stream) {
    const float* x   = (const float*)d_in[0];
    const float* sdf = (const float*)d_in[1];
    const int*   ei  = (const int*)d_in[2];
    const float* cx  = (const float*)d_in[3];
    const float* cy  = (const float*)d_in[4];
    const float* W[6]    = { (const float*)d_in[5],  (const float*)d_in[7],
                             (const float*)d_in[9],  (const float*)d_in[11],
                             (const float*)d_in[13], (const float*)d_in[15] };
    const float* bias[6] = { (const float*)d_in[6],  (const float*)d_in[8],
                             (const float*)d_in[10], (const float*)d_in[12],
                             (const float*)d_in[14], (const float*)d_in[16] };
    const int* row = ei;
    const int* col = ei + NEDGE;
    const int n = NTOT;

    char* ws = (char*)d_ws;
    size_t off = 0;
    auto carve = [&](size_t bytes) -> char* {
        char* p = ws + off;
        off += (bytes + 255) & ~(size_t)255;
        return p;
    };
    float*  deg  = (float*)carve((size_t)n * 4);
    float*  dinv = (float*)carve((size_t)n * 4);
    __bf16* bufA = (__bf16*)carve((size_t)n * KENDP * 2);  // ping
    __bf16* bufB = (__bf16*)carve((size_t)n * KENDP * 2);  // pong (also concat layout)
    float*  hbuf = (float*)carve((size_t)n * HDIM * 4);
    float*  accb = (float*)carve((size_t)n * HDIM * 4);
    float*  fy   = (float*)carve((size_t)n * OUTC * 4);
    __bf16* W0b  = (__bf16*)carve((size_t)K0PAD * HDIM * 2);
    __bf16* W1b  = (__bf16*)carve((size_t)HDIM * HDIM * 2);
    __bf16* W2b  = (__bf16*)carve((size_t)HDIM * HDIM * 2);
    __bf16* E0b  = (__bf16*)carve((size_t)KENDP * HDIM * 2);
    __bf16* E1b  = (__bf16*)carve((size_t)HDIM * HDIM * 2);
    __bf16* E2b  = (__bf16*)carve((size_t)HDIM * N2PAD * 2);

    auto cdiv = [](long long a, long long b) { return (int)((a + b - 1) / b); };

    // normalization
    k_deg_init <<<cdiv(n, 256), 256, 0, stream>>>(deg, n);
    k_deg_edges<<<cdiv(NEDGE, 256), 256, 0, stream>>>(col, deg, NEDGE);
    k_dinv     <<<cdiv(n, 256), 256, 0, stream>>>(deg, dinv, n);

    // bf16 prep (weights packed into WMMA fragment layout)
    k_build_in0<<<cdiv(n, 256), 256, 0, stream>>>(x, sdf, bufA, n);
    k_pack_b<<<cdiv((long long)K0PAD * HDIM, 256), 256, 0, stream>>>(W[0], W0b, DIN + 1, HDIM, K0PAD, HDIM);
    k_pack_b<<<cdiv((long long)HDIM * HDIM, 256), 256, 0, stream>>>(W[1], W1b, HDIM, HDIM, HDIM, HDIM);
    k_pack_b<<<cdiv((long long)HDIM * HDIM, 256), 256, 0, stream>>>(W[2], W2b, HDIM, HDIM, HDIM, HDIM);
    k_pack_b<<<cdiv((long long)KENDP * HDIM, 256), 256, 0, stream>>>(W[3], E0b, OUTC + HDIM, HDIM, KENDP, HDIM);
    k_pack_b<<<cdiv((long long)HDIM * HDIM, 256), 256, 0, stream>>>(W[4], E1b, HDIM, HDIM, HDIM, HDIM);
    k_pack_b<<<cdiv((long long)HDIM * N2PAD, 256), 256, 0, stream>>>(W[5], E2b, HDIM, OUTC, HDIM, N2PAD);

    // 8 waves/block x 4 tiles x 16 cols = 512 cols per block -> grid.y == 1
    const dim3 gemmGrid(n / 16, 1);
    const int C4 = HDIM / 4;

    auto gcn_layer = [&](const __bf16* Ain, int Kpad, int lda, const __bf16* Wb,
                         const float* bvec, __bf16* outbf, int ldo, int coff) {
        k_gemm_bf16<4><<<gemmGrid, 256, 0, stream>>>(Ain, Wb, hbuf, n, Kpad / 32, HDIM, lda, HDIM);
        k_self_loop<<<cdiv((long long)n * C4, 256), 256, 0, stream>>>(hbuf, dinv, accb, n, C4);
        k_scatter_edges<<<cdiv((long long)NEDGE * C4, 256), 256, 0, stream>>>(
            hbuf, row, col, dinv, accb, NEDGE, C4);
        k_bias_relu_bf16<<<cdiv((long long)n * HDIM, 256), 256, 0, stream>>>(
            accb, bvec, outbf, n, HDIM, ldo, coff);
    };

    // pre convs
    gcn_layer(bufA, K0PAD, K0PAD, W0b, bias[0], bufB, HDIM, 0);
    gcn_layer(bufB, HDIM, HDIM, W1b, bias[1], bufA, HDIM, 0);
    gcn_layer(bufA, HDIM, HDIM, W2b, bias[2], bufB, KENDP, OUTC);  // into concat layout

    // KNN upsample + concat
    k_knn<<<cdiv(n, 256), 256, 0, stream>>>(x, cx, cy, fy, n);
    k_concat_fy<<<cdiv(n, 256), 256, 0, stream>>>(fy, bufB, n);

    // end convs
    gcn_layer(bufB, KENDP, KENDP, E0b, bias[3], bufA, HDIM, 0);
    gcn_layer(bufA, HDIM, HDIM, E1b, bias[4], bufB, HDIM, 0);

    // last layer: 512 -> 3 (padded to 16 cols), no relu
    k_gemm_bf16<1><<<dim3(n / 16, 1), 32, 0, stream>>>(bufB, E2b, hbuf, n, HDIM / 32, N2PAD, HDIM, N2PAD);
    k_self_loop<<<cdiv((long long)n * (N2PAD / 4), 256), 256, 0, stream>>>(hbuf, dinv, accb, n, N2PAD / 4);
    k_scatter_edges<<<cdiv((long long)NEDGE * (N2PAD / 4), 256), 256, 0, stream>>>(
        hbuf, row, col, dinv, accb, NEDGE, N2PAD / 4);
    k_final<<<cdiv((long long)n * OUTC, 256), 256, 0, stream>>>(accb, bias[5], (float*)d_out, n);
}
